// BidirectionalMambaLayer_51994874086091
// MI455X (gfx1250) — matrix operand order repaired and checked by reading, not compiled, not run on hardware
//
#include <hip/hip_runtime.h>
#include <hip/hip_bf16.h>
#include <math.h>

// ---------------- model constants ----------------
#define B_SZ      2
#define L_SEQ     2048
#define M_TOT     (B_SZ * L_SEQ)      // 4096 rows
#define D_MODEL   512
#define D_STATE   64
#define D_INNER   1024
#define HEADDIM   64
#define NHEADS    16
#define D_CONV    4
#define CONV_DIM  (D_INNER + 2 * D_STATE)               // 1152
#define D_IN_PROJ (2 * D_INNER + 2 * D_STATE + NHEADS)  // 2192
#define D_FFN     2048
#define RMS_EPS   1.1920929e-07f
#define GNORM_EPS 1e-05f

typedef __attribute__((ext_vector_type(16))) _Float16 v16h;
typedef __attribute__((ext_vector_type(8)))  _Float16 h8;
typedef __attribute__((ext_vector_type(8)))  float    v8f;
typedef __attribute__((ext_vector_type(4)))  unsigned int u32x4;
typedef __attribute__((ext_vector_type(8)))  int      i32x8;
typedef __attribute__((ext_vector_type(4)))  int      i32x4;

#ifndef __has_builtin
#define __has_builtin(x) 0
#endif
#if __has_builtin(__builtin_amdgcn_tensor_load_to_lds) && __has_builtin(__builtin_amdgcn_s_wait_tensorcnt)
#define HAS_TDM 1
#else
#define HAS_TDM 0
#endif

// ---------------- small helpers ----------------
__device__ __forceinline__ float silu_f(float x) { return x / (1.0f + __expf(-x)); }
__device__ __forceinline__ float softplus_f(float x) {
    return (x > 20.0f) ? x : log1pf(__expf(x));
}
__device__ __forceinline__ float gelu_exact(float x) {
    return 0.5f * x * (1.0f + erff(x * 0.70710678118654752f));
}

#if HAS_TDM
// Issue one 2D TDM tile load: global (row-major, f16 elements) -> LDS.
// D# packing per CDNA5 ISA ch.8: group0 = {count/type/addrs}, group1 = {dims/strides}.
__device__ __forceinline__ void tdm_load_2d_f16(unsigned lds_off, const void* gptr,
                                                unsigned tile_k, unsigned tile_rows,
                                                unsigned long long row_stride_elems) {
    unsigned long long ga = (unsigned long long)(size_t)gptr;
    u32x4 g0;
    g0[0] = 1u;                                            // count=1, user descriptor
    g0[1] = lds_off;                                       // lds_addr [63:32]
    g0[2] = (unsigned)(ga & 0xffffffffu);                  // global_addr[31:0]
    g0[3] = (unsigned)((ga >> 32) & 0x01ffffffu) | (2u << 30);  // global_addr[56:32] | type=2
    unsigned td0 = tile_k;        // tensor_dim == tile (tile always fully in-bounds)
    unsigned td1 = tile_rows;
    unsigned long long s0 = row_stride_elems;
    i32x8 g1;
    g1[0] = (int)(1u << 16);                               // wg_mask=0, data_size=1 (2 bytes)
    g1[1] = (int)((td0 & 0xffffu) << 16);                  // [63:48] = tensor_dim0[15:0]
    g1[2] = (int)(((td0 >> 16) & 0xffffu) | ((td1 & 0xffffu) << 16));
    g1[3] = (int)(((td1 >> 16) & 0xffffu) | ((tile_k & 0xffffu) << 16));   // tile_dim0
    g1[4] = (int)(tile_rows & 0xffffu);                    // tile_dim1 ; tile_dim2=0
    g1[5] = (int)(s0 & 0xffffffffu);                       // tensor_dim0_stride[31:0]
    g1[6] = (int)((s0 >> 32) & 0xffffu);                   // stride[47:32]; dim1_stride[15:0]=0
    g1[7] = 0;
    i32x4 z4 = {0, 0, 0, 0};
#if __clang_major__ >= 23
    i32x8 z8 = {0, 0, 0, 0, 0, 0, 0, 0};
    __builtin_amdgcn_tensor_load_to_lds(g0, g1, z4, z4, z8, 0);
#else
    __builtin_amdgcn_tensor_load_to_lds(g0, g1, z4, z4, 0);
#endif
}
#endif

// ---------------- f32 -> f16 convert ----------------
__global__ void cvt_f32_f16_kernel(const float* __restrict__ src, _Float16* __restrict__ dst, int n) {
    int i = blockIdx.x * blockDim.x + threadIdx.x;
    if (i < n) dst[i] = (_Float16)src[i];
}

// ---------------- RMS(x)*w + pos -> f16 (normal + time-flipped) ----------------
__global__ void rms_pos_kernel(const float* __restrict__ x, const float* __restrict__ pos,
                               const float* __restrict__ w,
                               _Float16* __restrict__ si_h, _Float16* __restrict__ si_rev_h) {
    int row = blockIdx.x;
    int t   = threadIdx.x;          // 256 threads, 2 cols each
    __shared__ float red[256];
    size_t base = (size_t)row * D_MODEL;
    float v0 = x[base + t];
    float v1 = x[base + 256 + t];
    red[t] = v0 * v0 + v1 * v1;
    __syncthreads();
    for (int s = 128; s > 0; s >>= 1) {
        if (t < s) red[t] += red[t + s];
        __syncthreads();
    }
    float r = rsqrtf(red[0] * (1.0f / D_MODEL) + RMS_EPS);
    int b = row / L_SEQ, l = row % L_SEQ;
    size_t rbase = ((size_t)b * L_SEQ + (L_SEQ - 1 - l)) * D_MODEL;
    float o0 = v0 * r * w[t]       + pos[base + t];
    float o1 = v1 * r * w[256 + t] + pos[base + 256 + t];
    si_h[base + t]            = (_Float16)o0;
    si_h[base + 256 + t]      = (_Float16)o1;
    si_rev_h[rbase + t]       = (_Float16)o0;
    si_rev_h[rbase + 256 + t] = (_Float16)o1;
}

// ---------------- WMMA GEMM, double-buffered TDM->LDS pipeline ----------------
// out(M,N) = A(M,K) * W(N,K)^T.  256 threads = 8 waves; block tile 256(M) x 16(N);
// each wave owns two 16x16 output tiles (m-tiles w and w+8) sharing one B fragment.
// Wave 0 drives the DMA: pre-issue tile 0, then per K-step issue the next tile into
// the ping-pong buffer and wait TENSORcnt<=2 (tensor ops retire in order, so the
// current tile's two descriptors are complete while the next tile streams in).
#define GEMM_BM 256
#define GEMM_BK 32
__global__ __launch_bounds__(256) void gemm_wmma_kernel(const _Float16* __restrict__ A,
                                                        const _Float16* __restrict__ W,
                                                        float* __restrict__ out,
                                                        int M, int N, int K, int Lseq,
                                                        int reverseRows, int accumulate) {
    __shared__ _Float16 smA[2][GEMM_BM * GEMM_BK];   // 2 x 16 KB
    __shared__ _Float16 smB[2][16 * GEMM_BK];        // 2 x 1 KB
    int tileN = blockIdx.x;                          // 16-wide
    int tileM = blockIdx.y;                          // 256-wide
    int t    = threadIdx.x;
    int wave = t >> 5;
    int lane = t & 31;
    int m16  = lane & 15;
    int hi   = lane >> 4;
    v8f acc0 = {}, acc1 = {};
    const _Float16* aTile0 = A + (size_t)tileM * GEMM_BM * K;
    const _Float16* bTile0 = W + (size_t)tileN * 16 * K;
    int nIter = K / GEMM_BK;
#if HAS_TDM
    unsigned ldsA[2] = { (unsigned)(size_t)(void*)&smA[0][0], (unsigned)(size_t)(void*)&smA[1][0] };
    unsigned ldsB[2] = { (unsigned)(size_t)(void*)&smB[0][0], (unsigned)(size_t)(void*)&smB[1][0] };
    if (t < 32) {   // prologue: fill buffer 0
        tdm_load_2d_f16(ldsA[0], aTile0, GEMM_BK, GEMM_BM, (unsigned long long)K);
        tdm_load_2d_f16(ldsB[0], bTile0, GEMM_BK, 16,      (unsigned long long)K);
    }
#endif
    for (int it = 0; it < nIter; ++it) {
        int k0  = it * GEMM_BK;
        int cur = it & 1;
#if HAS_TDM
        if (t < 32) {
            if (it + 1 < nIter) {
                tdm_load_2d_f16(ldsA[cur ^ 1], aTile0 + k0 + GEMM_BK, GEMM_BK, GEMM_BM,
                                (unsigned long long)K);
                tdm_load_2d_f16(ldsB[cur ^ 1], bTile0 + k0 + GEMM_BK, GEMM_BK, 16,
                                (unsigned long long)K);
                __builtin_amdgcn_s_wait_tensorcnt(2);   // current tile done; next in flight
            } else {
                __builtin_amdgcn_s_wait_tensorcnt(0);
            }
        }
        __syncthreads();
#else
        // cooperative staging fallback (8 halves per 128-bit copy)
        for (int i = t; i < (GEMM_BM * GEMM_BK) / 8; i += 256) {
            int r = i >> 2, kk = (i & 3) * 8;
            *(h8*)&smA[cur][r * GEMM_BK + kk] = *(const h8*)(aTile0 + (size_t)r * K + k0 + kk);
        }
        for (int i = t; i < (16 * GEMM_BK) / 8; i += 256) {
            int r = i >> 2, kk = (i & 3) * 8;
            *(h8*)&smB[cur][r * GEMM_BK + kk] = *(const h8*)(bTile0 + (size_t)r * K + k0 + kk);
        }
        __syncthreads();
#endif
        // A frag: lane m = l&15, K halves {base..base+7, base+16..base+23}, base = 8*(l>>4)
        const _Float16* aL0 = &smA[cur][(wave * 16 + m16) * GEMM_BK + hi * 8];
        const _Float16* aL1 = aL0 + 128 * GEMM_BK;           // m-tile (wave+8)
        const _Float16* bL  = &smB[cur][m16 * GEMM_BK + hi * 8];
        h8 a00 = *(const h8*)(aL0);
        h8 a01 = *(const h8*)(aL0 + 16);
        h8 a10 = *(const h8*)(aL1);
        h8 a11 = *(const h8*)(aL1 + 16);
        h8 b0  = *(const h8*)(bL);
        h8 b1  = *(const h8*)(bL + 16);
        v16h av0, av1, bv;
        #pragma unroll
        for (int i = 0; i < 8; ++i) {
            av0[i] = a00[i]; av0[i + 8] = a01[i];
            av1[i] = a10[i]; av1[i + 8] = a11[i];
            bv[i]  = b0[i];  bv[i + 8]  = b1[i];
        }
        acc0 = __builtin_amdgcn_wmma_f32_16x16x32_f16(false, av0, false, bv, (short)0, acc0, false, false);
        acc1 = __builtin_amdgcn_wmma_f32_16x16x32_f16(false, av1, false, bv, (short)0, acc1, false, false);
        __syncthreads();   // all reads of buffer `cur` done before it is refilled at it+2
    }
    int ncol = tileN * 16 + m16;
    #pragma unroll
    for (int v = 0; v < 8; ++v) {
        int row0 = tileM * GEMM_BM + wave * 16 + v + 8 * hi;  // D layout: M = v + 8*(lane>>4)
        #pragma unroll
        for (int half = 0; half < 2; ++half) {
            int row = row0 + half * 128;
            float val = half ? acc1[v] : acc0[v];
            int orow = row;
            if (reverseRows) {
                int b = row / Lseq, l = row % Lseq;
                orow = b * Lseq + (Lseq - 1 - l);
            }
            size_t idx = (size_t)orow * N + ncol;
            if (accumulate) out[idx] += val; else out[idx] = val;
        }
    }
}

// ---------------- dt = softplus(raw + bias); dA = exp(-exp(A_log)*dt) ----------------
__global__ void dt_kernel(const float* __restrict__ zx, const float* __restrict__ dt_bias,
                          const float* __restrict__ A_log,
                          float* __restrict__ dt_s, float* __restrict__ dA_s) {
    int idx = blockIdx.x * blockDim.x + threadIdx.x;   // M_TOT * NHEADS
    if (idx >= M_TOT * NHEADS) return;
    int h = idx & (NHEADS - 1);
    int row = idx >> 4;
    float raw = zx[(size_t)row * D_IN_PROJ + (2 * D_INNER + 2 * D_STATE) + h] + dt_bias[h];
    float dt = softplus_f(raw);
    dt_s[idx] = dt;
    dA_s[idx] = __expf(-__expf(A_log[h]) * dt);
}

// ---------------- causal depthwise conv(4) + silu ----------------
__global__ void conv_kernel(const float* __restrict__ zx, const float* __restrict__ cw,
                            const float* __restrict__ cb, float* __restrict__ xBC) {
    size_t idx = (size_t)blockIdx.x * blockDim.x + threadIdx.x;   // M_TOT * CONV_DIM
    if (idx >= (size_t)M_TOT * CONV_DIM) return;
    int c   = (int)(idx % CONV_DIM);
    int row = (int)(idx / CONV_DIM);
    int b = row / L_SEQ, l = row % L_SEQ;
    float acc = cb[c];
    #pragma unroll
    for (int i = 0; i < D_CONV; ++i) {
        int ls = l - (D_CONV - 1) + i;
        if (ls >= 0)
            acc += zx[((size_t)b * L_SEQ + ls) * D_IN_PROJ + D_INNER + c] * cw[c * D_CONV + i];
    }
    xBC[idx] = silu_f(acc);
}

// ---------------- sequential SSM scan; 64x64 state in registers ----------------
// one block per (batch, head); 256 threads; thread t owns p = t>>2, n in [(t&3)*16, +16)
__global__ void scan_kernel(const float* __restrict__ xBC, const float* __restrict__ dt_s,
                            const float* __restrict__ dA_s, const float* __restrict__ Dp,
                            float* __restrict__ y) {
    int blk = blockIdx.x;                 // 0..31
    int b = blk / NHEADS, h = blk % NHEADS;
    int t = threadIdx.x;
    int p  = t >> 2;
    int n0 = (t & 3) * 16;
    float hst[16];
    #pragma unroll
    for (int i = 0; i < 16; ++i) hst[i] = 0.0f;
    float Dh = Dp[h];
    for (int l = 0; l < L_SEQ; ++l) {
        size_t row = (size_t)b * L_SEQ + l;
        const float* xrow = xBC + row * CONV_DIM;
        if (l + 8 < L_SEQ) {  // hide load latency of future timesteps (global_prefetch_b8)
            const float* frow = xrow + 8 * CONV_DIM;
            __builtin_prefetch(frow + h * HEADDIM + p, 0, 1);
            __builtin_prefetch(frow + D_INNER + n0, 0, 1);
            __builtin_prefetch(frow + D_INNER + D_STATE + n0, 0, 1);
        }
        float xv = xrow[h * HEADDIM + p];
        float dt = dt_s[row * NHEADS + h];
        float dA = dA_s[row * NHEADS + h];
        float4 Bq[4], Cq[4];
        #pragma unroll
        for (int q = 0; q < 4; ++q) {
            Bq[q] = *(const float4*)(xrow + D_INNER + n0 + 4 * q);
            Cq[q] = *(const float4*)(xrow + D_INNER + D_STATE + n0 + 4 * q);
        }
        float coef = dt * xv;
        float part = 0.0f;
        const float* Bv = (const float*)Bq;
        const float* Cv = (const float*)Cq;
        #pragma unroll
        for (int i = 0; i < 16; ++i) {
            hst[i] = hst[i] * dA + coef * Bv[i];
            part += hst[i] * Cv[i];
        }
        part += __shfl_xor(part, 1, 32);
        part += __shfl_xor(part, 2, 32);
        if ((t & 3) == 0)
            y[row * D_INNER + h * HEADDIM + p] = part + Dh * xv;
    }
}

// ---------------- y *= silu(z); groupnorm(RMS over 1024) -> f16 ----------------
__global__ void gate_gnorm_kernel(const float* __restrict__ y, const float* __restrict__ zx,
                                  const float* __restrict__ gw, _Float16* __restrict__ yh) {
    int row = blockIdx.x;
    int t   = threadIdx.x;          // 256 threads, 4 cols each
    __shared__ float red[256];
    float g[4];
    float ss = 0.0f;
    #pragma unroll
    for (int i = 0; i < 4; ++i) {
        int c = t * 4 + i;
        float z  = zx[(size_t)row * D_IN_PROJ + c];
        float gv = y[(size_t)row * D_INNER + c] * silu_f(z);
        g[i] = gv;
        ss += gv * gv;
    }
    red[t] = ss;
    __syncthreads();
    for (int s = 128; s > 0; s >>= 1) {
        if (t < s) red[t] += red[t + s];
        __syncthreads();
    }
    float r = rsqrtf(red[0] * (1.0f / D_INNER) + GNORM_EPS);
    #pragma unroll
    for (int i = 0; i < 4; ++i) {
        int c = t * 4 + i;
        yh[(size_t)row * D_INNER + c] = (_Float16)(g[i] * r * gw[c]);
    }
}

// ---------------- x1 = x + ssm_sum * pixel_mask ----------------
__global__ void resid_mask_kernel(const float* __restrict__ x, const float* __restrict__ ssm,
                                  const float* __restrict__ mask, float* __restrict__ x1) {
    size_t idx = (size_t)blockIdx.x * blockDim.x + threadIdx.x;
    if (idx >= (size_t)M_TOT * D_MODEL) return;
    int row = (int)(idx / D_MODEL);
    x1[idx] = x[idx] + ssm[idx] * mask[row];
}

// ---------------- RMS(x1)*w -> f16 (FFN input) ----------------
__global__ void rms_ffn_kernel(const float* __restrict__ x1, const float* __restrict__ w,
                               _Float16* __restrict__ hh) {
    int row = blockIdx.x;
    int t   = threadIdx.x;
    __shared__ float red[256];
    size_t base = (size_t)row * D_MODEL;
    float v0 = x1[base + t];
    float v1 = x1[base + 256 + t];
    red[t] = v0 * v0 + v1 * v1;
    __syncthreads();
    for (int s = 128; s > 0; s >>= 1) {
        if (t < s) red[t] += red[t + s];
        __syncthreads();
    }
    float r = rsqrtf(red[0] * (1.0f / D_MODEL) + RMS_EPS);
    hh[base + t]       = (_Float16)(v0 * r * w[t]);
    hh[base + 256 + t] = (_Float16)(v1 * r * w[256 + t]);
}

// ---------------- h = gelu(h_raw + b1) -> f16 ----------------
__global__ void bias_gelu_kernel(const float* __restrict__ hraw, const float* __restrict__ b1,
                                 _Float16* __restrict__ hh) {
    size_t idx = (size_t)blockIdx.x * blockDim.x + threadIdx.x;
    if (idx >= (size_t)M_TOT * D_FFN) return;
    int c = (int)(idx % D_FFN);
    hh[idx] = (_Float16)gelu_exact(hraw[idx] + b1[c]);
}

// ---------------- d_out = x1 + b2 (FFN2 GEMM then accumulates) ----------------
__global__ void out_init_kernel(const float* __restrict__ x1, const float* __restrict__ b2,
                                float* __restrict__ out) {
    size_t idx = (size_t)blockIdx.x * blockDim.x + threadIdx.x;
    if (idx >= (size_t)M_TOT * D_MODEL) return;
    int c = (int)(idx % D_MODEL);
    out[idx] = x1[idx] + b2[c];
}

// ================= host-side orchestration =================
extern "C" void kernel_launch(void* const* d_in, const int* in_sizes, int n_in,
                              void* d_out, int out_size, void* d_ws, size_t ws_size,
                              hipStream_t stream) {
    const float* x          = (const float*)d_in[0];
    const float* pos_emb    = (const float*)d_in[1];
    const float* pixel_mask = (const float*)d_in[2];
    const float* norm_ssm_w = (const float*)d_in[3];
    const float* norm_ffn_w = (const float*)d_in[4];
    const float* ffn_w1     = (const float*)d_in[5];
    const float* ffn_b1     = (const float*)d_in[6];
    const float* ffn_w2     = (const float*)d_in[7];
    const float* ffn_b2     = (const float*)d_in[8];
    const float* in_w[2]    = { (const float*)d_in[9],  (const float*)d_in[17] };
    const float* conv_w[2]  = { (const float*)d_in[10], (const float*)d_in[18] };
    const float* conv_b[2]  = { (const float*)d_in[11], (const float*)d_in[19] };
    const float* dt_bias[2] = { (const float*)d_in[12], (const float*)d_in[20] };
    const float* A_log[2]   = { (const float*)d_in[13], (const float*)d_in[21] };
    const float* Dp[2]      = { (const float*)d_in[14], (const float*)d_in[22] };
    const float* gnorm_w[2] = { (const float*)d_in[15], (const float*)d_in[23] };
    const float* out_w[2]   = { (const float*)d_in[16], (const float*)d_in[24] };
    float* out = (float*)d_out;

    // -------- workspace layout (bytes) --------
    char* ws = (char*)d_ws;
    size_t off = 0;
    auto alloc = [&](size_t bytes) { size_t o = off; off += (bytes + 255) & ~(size_t)255; return o; };
    size_t OFF_SI_H  = alloc((size_t)M_TOT * D_MODEL * 2);
    size_t OFF_SI_RH = alloc((size_t)M_TOT * D_MODEL * 2);
    size_t OFF_W_IN[2]  = { alloc((size_t)D_IN_PROJ * D_MODEL * 2), alloc((size_t)D_IN_PROJ * D_MODEL * 2) };
    size_t OFF_W_OUT[2] = { alloc((size_t)D_MODEL * D_INNER * 2),   alloc((size_t)D_MODEL * D_INNER * 2) };
    size_t OFF_W_F1  = alloc((size_t)D_FFN * D_MODEL * 2);
    size_t OFF_W_F2  = alloc((size_t)D_MODEL * D_FFN * 2);
    size_t OFF_ZX    = alloc((size_t)M_TOT * D_IN_PROJ * 4);   // reused as FFN h_raw (smaller)
    size_t OFF_XBC   = alloc((size_t)M_TOT * CONV_DIM * 4);    // reused as FFN h f16 (smaller)
    size_t OFF_DT    = alloc((size_t)M_TOT * NHEADS * 4);
    size_t OFF_DA    = alloc((size_t)M_TOT * NHEADS * 4);
    size_t OFF_Y     = alloc((size_t)M_TOT * D_INNER * 4);
    size_t OFF_YH    = alloc((size_t)M_TOT * D_INNER * 2);
    size_t OFF_SSM   = alloc((size_t)M_TOT * D_MODEL * 4);
    size_t OFF_X1    = alloc((size_t)M_TOT * D_MODEL * 4);
    size_t OFF_HIN   = alloc((size_t)M_TOT * D_MODEL * 2);
    (void)ws_size;

    _Float16* si_h     = (_Float16*)(ws + OFF_SI_H);
    _Float16* si_rev_h = (_Float16*)(ws + OFF_SI_RH);
    _Float16* w_in_h[2]  = { (_Float16*)(ws + OFF_W_IN[0]),  (_Float16*)(ws + OFF_W_IN[1]) };
    _Float16* w_out_h[2] = { (_Float16*)(ws + OFF_W_OUT[0]), (_Float16*)(ws + OFF_W_OUT[1]) };
    _Float16* w_f1_h   = (_Float16*)(ws + OFF_W_F1);
    _Float16* w_f2_h   = (_Float16*)(ws + OFF_W_F2);
    float*    zx       = (float*)(ws + OFF_ZX);
    float*    xBC      = (float*)(ws + OFF_XBC);
    float*    dt_s     = (float*)(ws + OFF_DT);
    float*    dA_s     = (float*)(ws + OFF_DA);
    float*    ybuf     = (float*)(ws + OFF_Y);
    _Float16* y_h      = (_Float16*)(ws + OFF_YH);
    float*    ssm_sum  = (float*)(ws + OFF_SSM);
    float*    x1       = (float*)(ws + OFF_X1);
    _Float16* h_in_h   = (_Float16*)(ws + OFF_HIN);
    float*    h_raw    = zx;                        // reuse
    _Float16* h_h      = (_Float16*)(ws + OFF_XBC); // reuse

    const int T = 256;
    auto cvt = [&](const float* s, _Float16* d, int n) {
        cvt_f32_f16_kernel<<<(n + T - 1) / T, T, 0, stream>>>(s, d, n);
    };
    cvt(in_w[0],  w_in_h[0],  D_IN_PROJ * D_MODEL);
    cvt(in_w[1],  w_in_h[1],  D_IN_PROJ * D_MODEL);
    cvt(out_w[0], w_out_h[0], D_MODEL * D_INNER);
    cvt(out_w[1], w_out_h[1], D_MODEL * D_INNER);
    cvt(ffn_w1,   w_f1_h,     D_FFN * D_MODEL);
    cvt(ffn_w2,   w_f2_h,     D_MODEL * D_FFN);

    rms_pos_kernel<<<M_TOT, T, 0, stream>>>(x, pos_emb, norm_ssm_w, si_h, si_rev_h);

    dim3 blk(256, 1, 1);
    for (int dir = 0; dir < 2; ++dir) {
        const _Float16* si_d = (dir == 0) ? si_h : si_rev_h;
        // in_proj: (M,512) x (2192,512)^T
        gemm_wmma_kernel<<<dim3(D_IN_PROJ / 16, M_TOT / GEMM_BM), blk, 0, stream>>>(
            si_d, w_in_h[dir], zx, M_TOT, D_IN_PROJ, D_MODEL, L_SEQ, 0, 0);
        dt_kernel<<<(M_TOT * NHEADS + T - 1) / T, T, 0, stream>>>(
            zx, dt_bias[dir], A_log[dir], dt_s, dA_s);
        conv_kernel<<<(int)(((size_t)M_TOT * CONV_DIM + T - 1) / T), T, 0, stream>>>(
            zx, conv_w[dir], conv_b[dir], xBC);
        scan_kernel<<<B_SZ * NHEADS, T, 0, stream>>>(xBC, dt_s, dA_s, Dp[dir], ybuf);
        gate_gnorm_kernel<<<M_TOT, T, 0, stream>>>(ybuf, zx, gnorm_w[dir], y_h);
        // out_proj: (M,1024) x (512,1024)^T ; bwd un-flips rows and accumulates
        gemm_wmma_kernel<<<dim3(D_MODEL / 16, M_TOT / GEMM_BM), blk, 0, stream>>>(
            y_h, w_out_h[dir], ssm_sum, M_TOT, D_MODEL, D_INNER, L_SEQ,
            /*reverseRows=*/dir, /*accumulate=*/dir);
    }

    resid_mask_kernel<<<(M_TOT * D_MODEL + T - 1) / T, T, 0, stream>>>(x, ssm_sum, pixel_mask, x1);
    rms_ffn_kernel<<<M_TOT, T, 0, stream>>>(x1, norm_ffn_w, h_in_h);
    gemm_wmma_kernel<<<dim3(D_FFN / 16, M_TOT / GEMM_BM), blk, 0, stream>>>(
        h_in_h, w_f1_h, h_raw, M_TOT, D_FFN, D_MODEL, L_SEQ, 0, 0);
    bias_gelu_kernel<<<(int)(((size_t)M_TOT * D_FFN + T - 1) / T), T, 0, stream>>>(h_raw, ffn_b1, h_h);
    out_init_kernel<<<(M_TOT * D_MODEL + T - 1) / T, T, 0, stream>>>(x1, ffn_b2, out);
    gemm_wmma_kernel<<<dim3(D_MODEL / 16, M_TOT / GEMM_BM), blk, 0, stream>>>(
        h_h, w_f2_h, out, M_TOT, D_MODEL, D_FFN, L_SEQ, 0, /*accumulate=*/1);
}
